// UnifiedUnetGVPGNNModel_39608188404468
// MI455X (gfx1250) — matrix operand order repaired and checked by reading, not compiled, not run on hardware
//
#include <hip/hip_runtime.h>
#include <hip/hip_fp16.h>
#include <stdint.h>

// ---------------------------------------------------------------------------
// GVP-GNN U-Net for MI455X (gfx1250).
// Dense layers -> WMMA f16 (16x16x32, f32-accum) tiled GEMM with async
// global->LDS staging (ASYNCcnt). Irregular graph ops -> VALU/atomic kernels.
// ---------------------------------------------------------------------------

typedef _Float16 h16;
typedef __attribute__((ext_vector_type(4)))  _Float16 v4h;
typedef __attribute__((ext_vector_type(8)))  _Float16 v8h;
typedef __attribute__((ext_vector_type(16))) _Float16 v16h;
typedef __attribute__((ext_vector_type(8)))  float    v8f;

static __device__ __forceinline__ float wred32(float x) {
#pragma unroll
  for (int o = 16; o > 0; o >>= 1) x += __shfl_xor(x, o, 32);
  return x;
}

// gfx1250 async global->LDS copy (tracked by ASYNCcnt).
// Low 32 bits of a generic shared-memory pointer == LDS byte offset
// (LDS aperture tag lives in addr[63:32] per CDNA5 ISA 10.2).
static __device__ __forceinline__ void async_ld_b128(void* lds, const float* g) {
  uint32_t off = (uint32_t)(uintptr_t)lds;
  asm volatile("global_load_async_to_lds_b128 %0, %1, off"
               :: "v"(off), "v"(g) : "memory");
}
static __device__ __forceinline__ void wait_async0() {
  asm volatile("s_wait_asynccnt 0x0" ::: "memory");
}

// ---------------------------------------------------------------------------
// WMMA GEMM:  C[m,n] = act( sum_k A[m,k] * W[k,n] + bias[n] )
// A addressed as A[m*lar + k*lac]  (lac=1 contiguous, lac=3 for xyz slices)
// W addressed as W[k*ldw + n]      (row-major [K,N] weights)
// C addressed as C[m*lcr + n*lcc]
// Block: 256 threads = 8 waves (4x2); wave tile 32x32 = 2x2 WMMA 16x16 tiles.
// Interior tiles with 16B-aligned rows use GLOBAL_LOAD_ASYNC_TO_LDS_B128.
// ---------------------------------------------------------------------------
template <int ACT>
__global__ __launch_bounds__(256)
void wmma_gemm_k(const float* __restrict__ A, long lar, long lac,
                 const float* __restrict__ W, long ldw,
                 const float* __restrict__ bias,
                 float* __restrict__ C, long lcr, long lcc,
                 int M, int N, int K) {
  __shared__ h16   As[128][32];
  __shared__ h16   Bs[64][32];     // transposed: Bs[n][k]
  __shared__ float As32[128][32];  // async staging (f32)
  __shared__ float Bs32[32][64];   // async staging (f32), natural [k][n]
  const int tid  = threadIdx.x;
  const int lane = tid & 31;
  const int wave = tid >> 5;
  const int wm = wave >> 1, wn = wave & 1;
  const int bm = blockIdx.x * 128, bn = blockIdx.y * 64;
  const int kg = lane >> 4;
  const int mr = lane & 15;

  v8f acc[2][2] = {};

  for (int k0 = 0; k0 < K; k0 += 32) {
    const bool fastA = (lac == 1) && ((lar & 3) == 0) &&
                       (bm + 128 <= M) && (k0 + 32 <= K);
    const bool fastB = ((ldw & 3) == 0) && (bn + 64 <= N) && (k0 + 32 <= K);

    if (fastA) {
      // 128x32 f32 tile, 16B per async transfer
      for (int s = tid; s < 1024; s += 256) {
        int m = s >> 3, c4 = (s & 7) << 2;
        async_ld_b128(&As32[m][c4], A + (long)(bm + m) * lar + (k0 + c4));
      }
    } else {
      float tmp[16];
#pragma unroll
      for (int t = 0; t < 16; ++t) {
        int i = tid + t * 256;
        int m = i >> 5, k = i & 31;
        int gm = bm + m, gk = k0 + k;
        tmp[t] = (gm < M && gk < K) ? A[(long)gm * lar + (long)gk * lac] : 0.f;
      }
#pragma unroll
      for (int t = 0; t < 16; ++t) {
        int i = tid + t * 256;
        As[i >> 5][i & 31] = (h16)tmp[t];
      }
    }
    if (fastB) {
      for (int s = tid; s < 512; s += 256) {
        int k = s >> 4, n4 = (s & 15) << 2;
        async_ld_b128(&Bs32[k][n4], W + (long)(k0 + k) * ldw + (bn + n4));
      }
    } else {
      float tmp[8];
#pragma unroll
      for (int t = 0; t < 8; ++t) {
        int i = tid + t * 256;
        int n = i >> 5, k = i & 31;
        int gn = bn + n, gk = k0 + k;
        tmp[t] = (gn < N && gk < K) ? W[(long)gk * ldw + gn] : 0.f;
      }
#pragma unroll
      for (int t = 0; t < 8; ++t) {
        int i = tid + t * 256;
        Bs[i >> 5][i & 31] = (h16)tmp[t];
      }
    }
    if (fastA || fastB) wait_async0();
    __syncthreads();

    // convert async-staged f32 tiles to the f16 fragment tiles
    if (fastA) {
      for (int s = tid; s < 1024; s += 256) {
        int m = s >> 3, c4 = (s & 7) << 2;
        float4 f = *(const float4*)&As32[m][c4];
        v4h h;
        h[0] = (h16)f.x; h[1] = (h16)f.y; h[2] = (h16)f.z; h[3] = (h16)f.w;
        *(v4h*)&As[m][c4] = h;
      }
    }
    if (fastB) {
      for (int s = tid; s < 2048; s += 256) {
        int n = s >> 5, k = s & 31;
        Bs[n][k] = (h16)Bs32[k][n];
      }
    }
    __syncthreads();

    // prefetch next A slab (gfx1250 global_prefetch path)
    if (!fastA && k0 + 32 < K) {
      int m = tid >> 1, k = (tid & 1) * 16;
      int gm = bm + m;
      if (gm < M) __builtin_prefetch(&A[(long)gm * lar + (long)(k0 + 32 + k) * lac], 0, 1);
    }

    // fragments per CDNA5 ISA VGPR layouts
    v16h af[2], bf[2];
#pragma unroll
    for (int t = 0; t < 2; ++t) {
      const h16* ap = &As[wm * 32 + t * 16 + mr][kg * 8];
      v8h a0 = *(const v8h*)ap;          // K = kg*8 + 0..7
      v8h a1 = *(const v8h*)(ap + 16);   // K = 16 + kg*8 + 0..7
      af[t] = __builtin_shufflevector(a0, a1, 0, 1, 2, 3, 4, 5, 6, 7,
                                      8, 9, 10, 11, 12, 13, 14, 15);
      const h16* bp = &Bs[wn * 32 + t * 16 + mr][kg * 8];
      v8h b0 = *(const v8h*)bp;          // K = kg*8 + 0..7
      v8h b1 = *(const v8h*)(bp + 16);   // K = 16 + kg*8 + 0..7
      bf[t] = __builtin_shufflevector(b0, b1, 0, 1, 2, 3, 4, 5, 6, 7,
                                      8, 9, 10, 11, 12, 13, 14, 15);
    }
#pragma unroll
    for (int tm = 0; tm < 2; ++tm)
#pragma unroll
      for (int tn = 0; tn < 2; ++tn)
        acc[tm][tn] = __builtin_amdgcn_wmma_f32_16x16x32_f16(
            false, af[tm], false, bf[tn], (short)0, acc[tm][tn], false, false);
    __syncthreads();
  }

#pragma unroll
  for (int tm = 0; tm < 2; ++tm)
#pragma unroll
    for (int tn = 0; tn < 2; ++tn)
#pragma unroll
      for (int r = 0; r < 8; ++r) {
        int m = bm + wm * 32 + tm * 16 + kg * 8 + r;
        int n = bn + wn * 32 + tn * 16 + mr;
        if (m < M && n < N) {
          float v = acc[tm][tn][r];
          if (bias) v += bias[n];
          if (ACT == 1) v = fmaxf(v, 0.f);
          C[(long)m * lcr + (long)n * lcc] = v;
        }
      }
}

// ---------------------------------------------------------------------------
// helpers
// ---------------------------------------------------------------------------
__global__ void zero_f_k(float* p, long n) {
  long i = blockIdx.x * 256L + threadIdx.x;
  if (i < n) p[i] = 0.f;
}
__global__ void zero_i_k(int* p, long n) {
  long i = blockIdx.x * 256L + threadIdx.x;
  if (i < n) p[i] = 0;
}
__global__ void copy_cols_k(float* __restrict__ dst, long ldd,
                            const float* __restrict__ src, long lds,
                            long M, int D) {
  long i = blockIdx.x * 256L + threadIdx.x;
  if (i >= M * D) return;
  long m = i / D; int c = (int)(i % D);
  dst[m * ldd + c] = src[m * lds + c];
}
// vn[m,h] = sqrt(clip(|vh[m,h,:]|^2, 1e-8)) written into dst[:, coff + h]
__global__ void vnorm_k(const float* __restrict__ vh, long M, int H,
                        float* __restrict__ dst, long ldd, long coff) {
  long i = blockIdx.x * 256L + threadIdx.x;
  if (i >= M * H) return;
  long m = i / H; int h = (int)(i % H);
  const float* p = vh + (m * H + h) * 3;
  float s = p[0] * p[0] + p[1] * p[1] + p[2] * p[2];
  dst[m * ldd + coff + h] = sqrtf(fmaxf(s, 1e-8f));
}
__global__ void gate_mul_k(float* __restrict__ v, const float* __restrict__ gate,
                           long M, int V) {
  long i = blockIdx.x * 256L + threadIdx.x;
  if (i >= M * V) return;
  long m = i / V; int o = (int)(i % V);
  float g = 1.f / (1.f + expf(-gate[m * V + o]));
  float* p = v + (m * (long)V + o) * 3;
  p[0] *= g; p[1] *= g; p[2] *= g;
}
// residual (+optional /cnt) + layer_norm_sv. s: 128 dims, wave per node.
__global__ __launch_bounds__(256)
void res_ln_sv_k(const float* __restrict__ s0, const float* __restrict__ s1,
                 const float* __restrict__ v0, const float* __restrict__ v1,
                 const float* __restrict__ cnt,
                 const float* __restrict__ g, const float* __restrict__ b,
                 float* __restrict__ so, float* __restrict__ vo,
                 int n, int vdim) {
  int gt = blockIdx.x * 256 + threadIdx.x;
  int wid = gt >> 5, lane = gt & 31;
  if (wid >= n) return;
  float inv = 1.f;
  if (cnt) inv = 1.f / fmaxf(cnt[wid], 1.f);
  const float* p0 = s0 + (long)wid * 128;
  float x[4]; float lsum = 0.f;
#pragma unroll
  for (int t = 0; t < 4; ++t) {
    float v = p0[lane * 4 + t];
    if (s1) v += s1[(long)wid * 128 + lane * 4 + t] * inv;
    x[t] = v; lsum += v;
  }
  float mu = wred32(lsum) * (1.f / 128.f);
  float lv = 0.f;
#pragma unroll
  for (int t = 0; t < 4; ++t) { float d = x[t] - mu; lv += d * d; }
  float rstd = 1.f / sqrtf(wred32(lv) * (1.f / 128.f) + 1e-5f);
#pragma unroll
  for (int t = 0; t < 4; ++t) {
    int c = lane * 4 + t;
    so[(long)wid * 128 + c] = (x[t] - mu) * rstd * g[c] + b[c];
  }
  if (vo && vdim > 0) {
    float vx = 0.f, vy = 0.f, vz = 0.f, sq = 0.f;
    if (lane < vdim) {
      const float* pv = v0 + ((long)wid * vdim + lane) * 3;
      vx = pv[0]; vy = pv[1]; vz = pv[2];
      if (v1) {
        const float* pa = v1 + ((long)wid * vdim + lane) * 3;
        vx += pa[0] * inv; vy += pa[1] * inv; vz += pa[2] * inv;
      }
      sq = fmaxf(vx * vx + vy * vy + vz * vz, 1e-8f);
    }
    float rms = sqrtf(wred32(sq) / (float)vdim);
    if (lane < vdim) {
      float r = 1.f / rms;
      float* po = vo + ((long)wid * vdim + lane) * 3;
      po[0] = vx * r; po[1] = vy * r; po[2] = vz * r;
    }
  }
}
// fused RBF + layer_norm_sv + tiny edge GVP (9->32, vector 1->1, vector_gate)
__global__ void edge_feats_k(const float* __restrict__ pos,
                             const int* __restrict__ src, const int* __restrict__ dst,
                             int E,
                             const float* wh, const float* ws_b, const float* ws_w,
                             const float* wsv_b, const float* wsv_w, const float* wv,
                             const float* ln_b, const float* ln_g,
                             float* __restrict__ e_s, float* __restrict__ e_v) {
  int e = blockIdx.x * 256 + threadIdx.x;
  if (e >= E) return;
  int a = src[e], c = dst[e];
  float vx = pos[a * 3 + 0] - pos[c * 3 + 0];
  float vy = pos[a * 3 + 1] - pos[c * 3 + 1];
  float vz = pos[a * 3 + 2] - pos[c * 3 + 2];
  float len = sqrtf(vx * vx + vy * vy + vz * vz);
  float x = len * 0.1f;
  float x5 = x * x * x * x * x;
  float env = 1.f - 21.f * x5 + 35.f * x5 * x - 15.f * x5 * x * x;
  env = (x < 1.f) ? env : 0.f;
  float invl = (len > 0.f) ? 1.f / len : 0.f;
  const float c0 = 0.4472135954999579f;  // sqrt(2/10)
  float s[8];
#pragma unroll
  for (int n = 1; n <= 8; ++n)
    s[n - 1] = c0 * sinf((float)n * 3.14159265358979323846f * len * 0.1f) * invl * env;
  float ux = vx * invl, uy = vy * invl, uz = vz * invl;
  float mu = 0.f;
#pragma unroll
  for (int i = 0; i < 8; ++i) mu += s[i];
  mu *= 0.125f;
  float var = 0.f;
#pragma unroll
  for (int i = 0; i < 8; ++i) { float d = s[i] - mu; var += d * d; }
  var *= 0.125f;
  float rstd = 1.f / sqrtf(var + 1e-5f);
#pragma unroll
  for (int i = 0; i < 8; ++i) s[i] = (s[i] - mu) * rstd * ln_g[i] + ln_b[i];
  float vn0 = sqrtf(fmaxf(ux * ux + uy * uy + uz * uz, 1e-8f));
  ux /= vn0; uy /= vn0; uz /= vn0;
  float w0 = wh[0];
  float hx = ux * w0, hy = uy * w0, hz = uz * w0;
  float vhn = sqrtf(fmaxf(hx * hx + hy * hy + hz * hz, 1e-8f));
  float so[32];
#pragma unroll
  for (int o = 0; o < 32; ++o) {
    float acc = ws_b[o];
    for (int i = 0; i < 8; ++i) acc += s[i] * ws_w[i * 32 + o];
    acc += vhn * ws_w[8 * 32 + o];
    so[o] = acc;
  }
  float w1 = wv[0];
  float ox = hx * w1, oy = hy * w1, oz = hz * w1;
  float gt = wsv_b[0];
#pragma unroll
  for (int o = 0; o < 32; ++o) gt += so[o] * wsv_w[o];
  float gs = 1.f / (1.f + expf(-gt));
  ox *= gs; oy *= gs; oz *= gs;
#pragma unroll
  for (int o = 0; o < 32; ++o) e_s[(long)e * 32 + o] = so[o];
  e_v[(long)e * 3 + 0] = ox; e_v[(long)e * 3 + 1] = oy; e_v[(long)e * 3 + 2] = oz;
}
// m_s = [hs[src](128)|e_s(32)|hs[dst](128)] into cat (ld 324),
// m_v = [hv[src](16)|e_v(1)|hv[dst](16)] into mvcat (ld 99)
__global__ void concat_msg_k(const float* __restrict__ hs, const float* __restrict__ hv,
                             const float* __restrict__ es, const float* __restrict__ ev,
                             const int* __restrict__ src, const int* __restrict__ dst,
                             float* __restrict__ cat, float* __restrict__ mvcat, long E) {
  long e = blockIdx.x * 256L + threadIdx.x;
  if (e >= E) return;
  int a = src[e], b = dst[e];
  float* cs = cat + e * 324;
  const float* ha = hs + (long)a * 128;
  const float* hb = hs + (long)b * 128;
  for (int i = 0; i < 128; ++i) cs[i] = ha[i];
  for (int i = 0; i < 32; ++i) cs[128 + i] = es[e * 32 + i];
  for (int i = 0; i < 128; ++i) cs[160 + i] = hb[i];
  float* cv = mvcat + e * 99;
  const float* va = hv + (long)a * 48;
  const float* vb = hv + (long)b * 48;
  for (int i = 0; i < 48; ++i) cv[i] = va[i];
  cv[48] = ev[e * 3]; cv[49] = ev[e * 3 + 1]; cv[50] = ev[e * 3 + 2];
  for (int i = 0; i < 48; ++i) cv[51 + i] = vb[i];
}
__global__ void aggregate_k(const float* __restrict__ ms, const float* __restrict__ mv,
                            const int* __restrict__ dst,
                            float* __restrict__ aggs, float* __restrict__ aggv,
                            float* __restrict__ cnt, long E) {
  long e = blockIdx.x * 256L + threadIdx.x;
  if (e >= E) return;
  int d = dst[e];
  for (int i = 0; i < 128; ++i) atomicAdd(&aggs[(long)d * 128 + i], ms[e * 128 + i]);
  for (int i = 0; i < 48; ++i)  atomicAdd(&aggv[(long)d * 48 + i], mv[e * 48 + i]);
  atomicAdd(&cnt[d], 1.f);
}
// farthest-point sampling; deterministic, single block
__global__ __launch_bounds__(1024)
void fps_k(const float* __restrict__ pos, int n, int k,
           int* __restrict__ idx, float* __restrict__ mind) {
  __shared__ float sd[1024];
  __shared__ int   si[1024];
  int tid = threadIdx.x;
  for (int i = tid; i < n; i += 1024) mind[i] = 1e30f;
  __syncthreads();
  int last = 0;
  for (int it = 0; it < k; ++it) {
    if (tid == 0) idx[it] = last;
    float lx = pos[last * 3 + 0], ly = pos[last * 3 + 1], lz = pos[last * 3 + 2];
    float bd = -1.f; int bi = 0x7fffffff;
    for (int i = tid; i < n; i += 1024) {
      float dx = pos[i * 3 + 0] - lx, dy = pos[i * 3 + 1] - ly, dz = pos[i * 3 + 2] - lz;
      float m = fminf(mind[i], dx * dx + dy * dy + dz * dz);
      mind[i] = m;
      if (m > bd) { bd = m; bi = i; }
    }
    sd[tid] = bd; si[tid] = bi;
    __syncthreads();
    for (int s = 512; s > 0; s >>= 1) {
      if (tid < s) {
        if (sd[tid + s] > sd[tid] || (sd[tid + s] == sd[tid] && si[tid + s] < si[tid])) {
          sd[tid] = sd[tid + s]; si[tid] = si[tid + s];
        }
      }
      __syncthreads();
    }
    last = si[0];
    __syncthreads();
  }
}
__global__ void knn_k(const float* __restrict__ pos, int n,
                      int* __restrict__ esrc, int* __restrict__ edst) {
  int i = blockIdx.x * 256 + threadIdx.x;
  if (i >= n) return;
  float bd[16]; int bi[16];
#pragma unroll
  for (int t = 0; t < 16; ++t) { bd[t] = 1e30f; bi[t] = 0; }
  float xi = pos[i * 3], yi = pos[i * 3 + 1], zi = pos[i * 3 + 2];
  for (int j = 0; j < n; ++j) {
    if (j == i) continue;
    float dx = pos[j * 3] - xi, dy = pos[j * 3 + 1] - yi, dz = pos[j * 3 + 2] - zi;
    float d = dx * dx + dy * dy + dz * dz;
    if (d < bd[15]) {
      int t = 15;
      while (t > 0 && bd[t - 1] > d) { bd[t] = bd[t - 1]; bi[t] = bi[t - 1]; --t; }
      bd[t] = d; bi[t] = j;
    }
  }
#pragma unroll
  for (int t = 0; t < 16; ++t) { esrc[i * 16 + t] = bi[t]; edst[i * 16 + t] = i; }
}
__global__ void mark_k(const int* __restrict__ idx, int nk, int* __restrict__ kept) {
  int j = blockIdx.x * 256 + threadIdx.x;
  if (j < nk) kept[idx[j]] = 1;
}
__global__ void gather_k(const float* __restrict__ pos, const float* __restrict__ hs,
                         const float* __restrict__ hv, const int* __restrict__ idx,
                         int nk, float* __restrict__ npos, float* __restrict__ ns,
                         float* __restrict__ nv) {
  int j = blockIdx.x * 256 + threadIdx.x;
  if (j >= nk) return;
  int o = idx[j];
  for (int c = 0; c < 3; ++c) npos[j * 3 + c] = pos[o * 3 + c];
  for (int i = 0; i < 128; ++i) ns[(long)j * 128 + i] = hs[(long)o * 128 + i];
  for (int i = 0; i < 48; ++i)  nv[(long)j * 48 + i] = hv[(long)o * 48 + i];
}
__global__ void scatter_k(const float* __restrict__ pos, const float* __restrict__ hs,
                          const float* __restrict__ hv, const int* __restrict__ kept,
                          int n_old, const float* __restrict__ npos,
                          float* __restrict__ ns, float* __restrict__ nv, int nk) {
  int p = blockIdx.x * 256 + threadIdx.x;
  if (p >= n_old || kept[p]) return;
  float px = pos[p * 3], py = pos[p * 3 + 1], pz = pos[p * 3 + 2];
  float best = 1e30f; int bj = 0;
  for (int j = 0; j < nk; ++j) {
    float dx = npos[j * 3] - px, dy = npos[j * 3 + 1] - py, dz = npos[j * 3 + 2] - pz;
    float d = dx * dx + dy * dy + dz * dz;
    if (d < best) { best = d; bj = j; }
  }
  for (int i = 0; i < 128; ++i) atomicAdd(&ns[(long)bj * 128 + i], hs[(long)p * 128 + i]);
  for (int i = 0; i < 48; ++i)  atomicAdd(&nv[(long)bj * 48 + i], hv[(long)p * 48 + i]);
}
__global__ void colsum_k(const float* __restrict__ A, int n, float* __restrict__ out) {
  int c = threadIdx.x;
  float s = 0.f;
  for (int i = 0; i < n; ++i) s += A[(long)i * 128 + c];
  out[c] = s;
}

// ---------------------------------------------------------------------------
// host orchestration
// ---------------------------------------------------------------------------
extern "C" void kernel_launch(void* const* d_in, const int* in_sizes, int n_in,
                              void* d_out, int out_size, void* d_ws, size_t ws_size,
                              hipStream_t stream) {
  (void)in_sizes; (void)n_in; (void)out_size; (void)ws_size;
  const long N0 = 16000, N1 = 9600, N2 = 5760;
  const long E0 = N0 * 16;

  const float* x      = (const float*)d_in[0];
  const float* pos_in = (const float*)d_in[1];
  const int*   eidx   = (const int*)d_in[2];
  // d_in[3] = batch (unused; single graph)

  // Params: jax pytree flatten (sorted dict keys). Leaves start at d_in[4]:
  //  0..15  W_e[0..1]: each {wh, ws.b, ws.w, wsv.b, wsv.w, wv, ln.b, ln.g}
  //  16..18 W_out: wh, ws.b, ws.w        19..20 W_out_ln: b, g
  //  21..22 W_v.ws: b, w                 23..24 W_v_ln: b, g
  //  25..26 emb_in: b, w
  //  27+34*L per layer: ff0(wh,wsb,wsw,wsvb,wsvw,wv) ff1(+6) ln0.b/g(+12/13)
  //          ln1.b/g(+14/15) msg0(+16..21) msg1(+22..27) msg2(+28..33)
  auto P = [&](int i) -> const float* { return (const float*)d_in[4 + i]; };

  // ---- workspace arena ----
  size_t off = 0;
  auto alloc_f = [&](long n) -> float* {
    float* p = (float*)((char*)d_ws + off);
    off = (off + (size_t)n * 4 + 255) & ~(size_t)255;
    return p;
  };
  auto alloc_i = [&](long n) -> int* {
    int* p = (int*)((char*)d_ws + off);
    off = (off + (size_t)n * 4 + 255) & ~(size_t)255;
    return p;
  };
  float* hsA  = alloc_f(N0 * 128);
  float* hsB  = alloc_f(N0 * 128);
  float* hvA  = alloc_f(N0 * 48);
  float* hvB  = alloc_f(N0 * 48);
  float* posA = alloc_f(N0 * 3);
  float* posB = alloc_f(N0 * 3);
  float* es   = alloc_f(E0 * 32);
  float* ev   = alloc_f(E0 * 3);
  float* cat  = alloc_f(E0 * 324);   // padded leading dim (multiple of 4)
  float* sA   = alloc_f(E0 * 128);
  float* sB   = alloc_f(E0 * 128);
  float* vhb  = alloc_f(E0 * 99);
  float* mvcat= alloc_f(E0 * 99);
  float* mvA  = alloc_f(E0 * 48);
  float* mvB  = alloc_f(E0 * 48);
  float* gateb= alloc_f(E0 * 16);
  float* aggS = alloc_f(N0 * 128);
  float* aggV = alloc_f(N0 * 48);
  float* cntb = alloc_f(N0);
  float* mindb= alloc_f(N0);
  int*   idxb  = alloc_i(N0);
  int*   keptb = alloc_i(N0);
  int*   esrcB = alloc_i(E0);
  int*   edstB = alloc_i(E0);

  auto g1 = [](long n) { return dim3((unsigned)((n + 255) / 256)); };

  auto gemm = [&](const float* A, long lar, long lac, const float* W, long ldw,
                  const float* bias, float* C, long lcr, long lcc,
                  int M, int N, int K, int act) {
    dim3 grd((unsigned)((M + 127) / 128), (unsigned)((N + 63) / 64));
    if (act)
      wmma_gemm_k<1><<<grd, 256, 0, stream>>>(A, lar, lac, W, ldw, bias, C, lcr, lcc, M, N, K);
    else
      wmma_gemm_k<0><<<grd, 256, 0, stream>>>(A, lar, lac, W, ldw, bias, C, lcr, lcc, M, N, K);
  };

  // Full GVP (vector_gate=True). If s_in==null, cat[:,0:si] is pre-filled.
  auto gvp = [&](const float* s_in, int si, const float* v_in, int vi,
                 int so_d, int vo_d, int hh,
                 const float* wh, const float* wsb, const float* wsw,
                 const float* wsvb, const float* wsvw, const float* wvw,
                 float* s_out, float* v_out, long M, bool act) {
    long ldc = (long)((si + hh + 3) & ~3);  // padded for 16B-aligned rows
    for (int c = 0; c < 3; ++c)
      gemm(v_in + c, 3L * vi, 3, wh, hh, nullptr, vhb + c, 3L * hh, 3, (int)M, hh, vi, 0);
    vnorm_k<<<g1(M * hh), 256, 0, stream>>>(vhb, M, hh, cat, ldc, si);
    if (s_in)
      copy_cols_k<<<g1(M * si), 256, 0, stream>>>(cat, ldc, s_in, si, M, si);
    gemm(cat, ldc, 1, wsw, so_d, wsb, s_out, so_d, 1, (int)M, so_d, si + hh, act ? 1 : 0);
    if (vo_d > 0) {
      for (int c = 0; c < 3; ++c)
        gemm(vhb + c, 3L * hh, 3, wvw, vo_d, nullptr, v_out + c, 3L * vo_d, 3, (int)M, vo_d, hh, 0);
      gemm(s_out, so_d, 1, wsvw, vo_d, wsvb, gateb, vo_d, 1, (int)M, vo_d, so_d, 0);
      gate_mul_k<<<g1(M * vo_d), 256, 0, stream>>>(v_out, gateb, M, vo_d);
    }
  };

  // ---- embedding: h_s = W_v( LN( x@emb + b ) );  h_v = 0 ----
  gemm(x, 64, 1, P(26), 128, P(25), sA, 128, 1, (int)N0, 128, 64, 0);
  res_ln_sv_k<<<g1(N0 * 32), 256, 0, stream>>>(sA, nullptr, nullptr, nullptr, nullptr,
                                               P(24), P(23), sB, nullptr, (int)N0, 0);
  gemm(sB, 128, 1, P(22), 128, P(21), hsA, 128, 1, (int)N0, 128, 128, 0);
  zero_f_k<<<g1(N0 * 48), 256, 0, stream>>>(hvA, N0 * 48);

  float *hs_cur = hsA, *hv_cur = hvA, *hs_tmp = hsB, *hv_tmp = hvB;
  const float* pos_cur = pos_in;
  int n_cur = (int)N0;
  long E = E0;
  const int* esrc = eidx;
  const int* edst = eidx + E0;

  // initial edge features with W_e[0]
  edge_feats_k<<<g1(E), 256, 0, stream>>>(pos_cur, esrc, edst, (int)E,
      P(0), P(1), P(2), P(3), P(4), P(5), P(6), P(7), es, ev);

  for (int li = 0; li < 5; ++li) {
    if (li == 1 || li == 3) {
      int j = (li == 1) ? 0 : 1;
      int nk = (li == 1) ? (int)N1 : (int)N2;
      float* pos_nxt = (li == 1) ? posA : posB;
      fps_k<<<1, 1024, 0, stream>>>(pos_cur, n_cur, nk, idxb, mindb);
      zero_i_k<<<g1(n_cur), 256, 0, stream>>>(keptb, n_cur);
      mark_k<<<g1(nk), 256, 0, stream>>>(idxb, nk, keptb);
      gather_k<<<g1(nk), 256, 0, stream>>>(pos_cur, hs_cur, hv_cur, idxb, nk,
                                           pos_nxt, hs_tmp, hv_tmp);
      scatter_k<<<g1(n_cur), 256, 0, stream>>>(pos_cur, hs_cur, hv_cur, keptb, n_cur,
                                               pos_nxt, hs_tmp, hv_tmp, nk);
      { float* t = hs_cur; hs_cur = hs_tmp; hs_tmp = t;
        t = hv_cur; hv_cur = hv_tmp; hv_tmp = t; }
      pos_cur = pos_nxt; n_cur = nk; E = (long)nk * 16;
      knn_k<<<g1(n_cur), 256, 0, stream>>>(pos_cur, n_cur, esrcB, edstB);
      esrc = esrcB; edst = edstB;
      int jb = j * 8;
      edge_feats_k<<<g1(E), 256, 0, stream>>>(pos_cur, esrc, edst, (int)E,
          P(jb + 0), P(jb + 1), P(jb + 2), P(jb + 3), P(jb + 4), P(jb + 5),
          P(jb + 6), P(jb + 7), es, ev);
    }

    const int base = 27 + 34 * li;
    // ---- message path (per edge) ----
    concat_msg_k<<<g1(E), 256, 0, stream>>>(hs_cur, hv_cur, es, ev, esrc, edst,
                                            cat, mvcat, E);
    gvp(nullptr, 288, mvcat, 33, 128, 16, 33,
        P(base + 16), P(base + 17), P(base + 18), P(base + 19), P(base + 20), P(base + 21),
        sA, mvA, E, true);
    gvp(sA, 128, mvA, 16, 128, 16, 16,
        P(base + 22), P(base + 23), P(base + 24), P(base + 25), P(base + 26), P(base + 27),
        sB, mvB, E, true);
    gvp(sB, 128, mvB, 16, 128, 16, 16,
        P(base + 28), P(base + 29), P(base + 30), P(base + 31), P(base + 32), P(base + 33),
        sA, mvA, E, false);
    // ---- mean aggregation + residual + ln0 ----
    zero_f_k<<<g1((long)n_cur * 128), 256, 0, stream>>>(aggS, (long)n_cur * 128);
    zero_f_k<<<g1((long)n_cur * 48), 256, 0, stream>>>(aggV, (long)n_cur * 48);
    zero_f_k<<<g1(n_cur), 256, 0, stream>>>(cntb, n_cur);
    aggregate_k<<<g1(E), 256, 0, stream>>>(sA, mvA, edst, aggS, aggV, cntb, E);
    res_ln_sv_k<<<g1((long)n_cur * 32), 256, 0, stream>>>(
        hs_cur, aggS, hv_cur, aggV, cntb, P(base + 13), P(base + 12),
        hs_tmp, hv_tmp, n_cur, 16);
    // ---- feed-forward (per node) + residual + ln1 ----
    gvp(hs_tmp, 128, hv_tmp, 16, 512, 32, 32,
        P(base + 0), P(base + 1), P(base + 2), P(base + 3), P(base + 4), P(base + 5),
        sA, mvA, n_cur, true);
    gvp(sA, 512, mvA, 32, 128, 16, 32,
        P(base + 6), P(base + 7), P(base + 8), P(base + 9), P(base + 10), P(base + 11),
        sB, mvB, n_cur, false);
    res_ln_sv_k<<<g1((long)n_cur * 32), 256, 0, stream>>>(
        hs_tmp, sB, hv_tmp, mvB, nullptr, P(base + 15), P(base + 14),
        hs_cur, hv_cur, n_cur, 16);
  }

  // ---- output head ----
  res_ln_sv_k<<<g1((long)n_cur * 32), 256, 0, stream>>>(
      hs_cur, nullptr, hv_cur, nullptr, nullptr, P(20), P(19),
      hs_tmp, hv_tmp, n_cur, 16);
  float* out = (float*)d_out;
  gvp(hs_tmp, 128, hv_tmp, 16, 128, 0, 16,
      P(16), P(17), P(18), nullptr, nullptr, nullptr,
      out, nullptr, n_cur, true);
  colsum_k<<<1, 128, 0, stream>>>(out, n_cur, out + (long)n_cur * 128);
}
